// LinearAttentionTransformer_2714419331384
// MI455X (gfx1250) — matrix operand order repaired and checked by reading, not compiled, not run on hardware
//
#include <hip/hip_runtime.h>
#include <hip/hip_bf16.h>
#include <cstddef>
#include <cstdint>

typedef __bf16 bf16_t;
typedef __attribute__((ext_vector_type(16))) __bf16 v16bf;
typedef __attribute__((ext_vector_type(8)))  __bf16 v8bf;
typedef __attribute__((ext_vector_type(8)))  float  v8f;
typedef unsigned int u32;
typedef __attribute__((ext_vector_type(4))) u32 v4u;
typedef __attribute__((ext_vector_type(8))) int v8i;
typedef __attribute__((ext_vector_type(4))) int v4i;

#if defined(__has_builtin)
#if __has_builtin(__builtin_amdgcn_tensor_load_to_lds)
#define HAVE_TDM 1
#endif
#endif

#define B_      2
#define T_      4096
#define DIM_    1024
#define DH_     128
#define MTOT    (B_ * T_)                    // 8192 tokens
#define NELEM   ((size_t)MTOT * DIM_)        // 8388608

// ---------------------------------------------------------------------------
// WMMA GEMM: C[M,N] = A[M,K] (bf16, row major) x B[K,N] (bf16, row major)
//            (+bias) (+gelu) -> f32 or bf16 out.
// Block: 256 threads = 8 waves. Macro tile 128x128, wave tile 32x64
// (2x4 tiles of v_wmma_f32_16x16x32_bf16), K-step 32.
// A tile staged by the Tensor Data Mover (TDM) with LDS row padding;
// B tile staged transposed by the threads (TDM cannot transpose).
// ---------------------------------------------------------------------------
#define GTM 128
#define GTN 128
#define GTK 32
#define GPAD 8   // LDS row pad (elements) -> row stride 80B, keeps 16B align

template <int ACT, int OUTBF16>   // ACT: 0=none 1=gelu(exact)
__global__ void __launch_bounds__(256)
gemm_bf16_wmma(const bf16_t* __restrict__ A, const bf16_t* __restrict__ B,
               const float* __restrict__ bias,
               float* __restrict__ Cf, bf16_t* __restrict__ Cb,
               int M, int N, int K)
{
    __shared__ __align__(16) bf16_t As[GTM][GTK + GPAD];
    __shared__ __align__(16) bf16_t Bs[GTN][GTK + GPAD];   // transposed: [n][k]

    const int tid  = threadIdx.x;
    const int wave = tid >> 5;
    const int lane = tid & 31;
    const int half = lane >> 4;      // 0/1
    const int r    = lane & 15;      // 0..15

    const int wm = wave & 3;         // wave m-offset = wm*32
    const int wn = wave >> 2;        // wave n-offset = wn*64

    const int gm0 = blockIdx.y * GTM;
    const int gn0 = blockIdx.x * GTN;

    v8f acc[2][4];
    #pragma unroll
    for (int i = 0; i < 2; ++i)
        #pragma unroll
        for (int j = 0; j < 4; ++j)
            #pragma unroll
            for (int v = 0; v < 8; ++v) acc[i][j][v] = 0.0f;

    // cooperative tile-load coordinates (B always; A only without TDM)
    const int bk   = tid & 31;            // 0..31
    const int bn0  = (tid >> 5) * 16;     // 0,16,...,112

#if HAVE_TDM
    // TDM descriptor pieces that do not depend on k0.
    const u32 ldsA = (u32)(uintptr_t)(&As[0][0]);   // generic LDS addr [31:0] = LDS offset
    // group1: workgroup_mask=0 | data_size=1(2B)<<16 | pad_enable<<20 |
    //         pad_interval=3 (64B of data)<<22 | pad_amount=3 (4 DWORDs=16B)<<25
    const u32 g1w0 = (1u << 16) | (1u << 20) | (3u << 22) | (3u << 25);
    v8i g1;
    g1[0] = (int)g1w0;
    g1[1] = (int)((u32)(GTK & 0xffff) << 16);               // tensor_dim0 lo16 @bits[63:48]
    g1[2] = (int)(((u32)GTK >> 16) | ((u32)(GTM & 0xffff) << 16)); // dim0 hi | tensor_dim1 lo16
    g1[3] = (int)(((u32)GTM >> 16) | ((u32)GTK << 16));     // tensor_dim1 hi | tile_dim0=32
    g1[4] = (int)((u32)GTM);                                // tile_dim1=128, tile_dim2=0
    g1[5] = (int)(u32)K;                                    // tensor_dim0_stride lo32 (elements)
    g1[6] = 0;                                              // stride hi16 | dim1_stride lo16
    g1[7] = 0;
    const v4i gz = {0, 0, 0, 0};
#if __clang_major__ >= 23
    const v8i gz8 = {0, 0, 0, 0, 0, 0, 0, 0};
#endif
#else
    const int arow = tid >> 1;            // 0..127
    const int acol = (tid & 1) * 16;      // 0 or 16
#endif

    for (int k0 = 0; k0 < K; k0 += GTK) {
        // ---- stage A ----
#if HAVE_TDM
        if (tid < 32) {   // one wave issues the tensor DMA (EXEC ignored by TDM)
            const unsigned long long ga =
                (unsigned long long)(uintptr_t)(A + (size_t)gm0 * K + k0);
            v4u g0;
            g0[0] = 1u;                                  // count=1 valid descriptor
            g0[1] = ldsA;                                // LDS byte address
            g0[2] = (u32)(ga & 0xffffffffu);             // global_addr lo
            g0[3] = (u32)((ga >> 32) & 0x01ffffffu) | (2u << 30);  // addr hi | type=2
#if __clang_major__ >= 23
            __builtin_amdgcn_tensor_load_to_lds(g0, g1, gz, gz, gz8, 0);
#else
            __builtin_amdgcn_tensor_load_to_lds(g0, g1, gz, gz, 0);
#endif
        }
#else
        const bf16_t* ag = A + (size_t)(gm0 + arow) * K + k0 + acol;
        v8bf a0 = *(const v8bf*)(ag);
        v8bf a1 = *(const v8bf*)(ag + 8);
        *(v8bf*)&As[arow][acol]     = a0;
        *(v8bf*)&As[arow][acol + 8] = a1;
#endif

        // ---- stage B transposed: Bs[n][k] ----
        const bf16_t* bg = B + (size_t)(k0 + bk) * N + gn0 + bn0;
        v8bf b0 = *(const v8bf*)(bg);
        v8bf b1 = *(const v8bf*)(bg + 8);
        #pragma unroll
        for (int i = 0; i < 8; ++i) Bs[bn0 + i][bk]     = b0[i];
        #pragma unroll
        for (int i = 0; i < 8; ++i) Bs[bn0 + 8 + i][bk] = b1[i];

        if (k0 + GTK < K) {   // global_prefetch_b8 of the next B K-tile
            __builtin_prefetch(B + (size_t)(k0 + GTK + bk) * N + gn0 + bn0, 0, 0);
        }

#if HAVE_TDM
        __builtin_amdgcn_s_wait_tensorcnt(0);   // TDM tile landed in LDS
#endif
        __syncthreads();

        // ---- A fragments: 16-bit A 16x32 layout:
        //  lane(half=0) holds row r, K = {0..7, 16..23}; half=1: K = {8..15, 24..31}
        v16bf afrag[2];
        #pragma unroll
        for (int mt = 0; mt < 2; ++mt) {
            const int row = wm * 32 + mt * 16 + r;
            v8bf lo = *(const v8bf*)&As[row][half * 8];
            v8bf hi = *(const v8bf*)&As[row][16 + half * 8];
            afrag[mt] = __builtin_shufflevector(lo, hi,
                0,1,2,3,4,5,6,7,8,9,10,11,12,13,14,15);
        }

        // ---- B fragments: B 32x16: lanes 0-15 K={0..7,16..23}, lanes 16-31 K={8..15,24..31},
        //  column n = lane%16 -> contiguous in transposed LDS tile.
        #pragma unroll
        for (int nt = 0; nt < 4; ++nt) {
            const int col = wn * 64 + nt * 16 + r;
            v8bf lo = *(const v8bf*)&Bs[col][half * 8];
            v8bf hi = *(const v8bf*)&Bs[col][16 + half * 8];
            v16bf bfrag = __builtin_shufflevector(lo, hi,
                0,1,2,3,4,5,6,7,8,9,10,11,12,13,14,15);
            #pragma unroll
            for (int mt = 0; mt < 2; ++mt) {
                acc[mt][nt] = __builtin_amdgcn_wmma_f32_16x16x32_bf16(
                    false, afrag[mt], false, bfrag,
                    (short)0, acc[mt][nt], false, false);
            }
        }
        __syncthreads();
    }

    // ---- epilogue: C/D layout: VGPR v -> M = v (lanes 0-15) / 8+v (lanes 16-31),
    //      N = lane%16
    #pragma unroll
    for (int mt = 0; mt < 2; ++mt) {
        const int rowb = gm0 + wm * 32 + mt * 16 + half * 8;
        #pragma unroll
        for (int nt = 0; nt < 4; ++nt) {
            const int col = gn0 + wn * 64 + nt * 16 + r;
            const float bv = bias ? bias[col] : 0.0f;
            #pragma unroll
            for (int v = 0; v < 8; ++v) {
                float x = acc[mt][nt][v] + bv;
                if (ACT == 1) x = 0.5f * x * (1.0f + erff(x * 0.70710678118654752f));
                const size_t idx = (size_t)(rowb + v) * N + col;
                if (OUTBF16) Cb[idx] = (bf16_t)x;
                else         Cf[idx] = x;
            }
        }
    }
}

// ---------------------------------------------------------------------------
// LayerNorm over DIM=1024, output bf16. One block per row.
// ---------------------------------------------------------------------------
__global__ void __launch_bounds__(256)
layernorm_to_bf16(const float* __restrict__ X, const float* __restrict__ g,
                  const float* __restrict__ b, bf16_t* __restrict__ out)
{
    __shared__ float red[256];
    const int tid = threadIdx.x;
    const size_t row = blockIdx.x;
    const float* xr = X + row * DIM_;

    float v[4];
    float s = 0.0f;
    #pragma unroll
    for (int i = 0; i < 4; ++i) { v[i] = xr[tid + i * 256]; s += v[i]; }
    red[tid] = s; __syncthreads();
    for (int st = 128; st > 0; st >>= 1) {
        if (tid < st) red[tid] += red[tid + st];
        __syncthreads();
    }
    const float mean = red[0] * (1.0f / DIM_);
    __syncthreads();

    s = 0.0f;
    #pragma unroll
    for (int i = 0; i < 4; ++i) { float d = v[i] - mean; s += d * d; }
    red[tid] = s; __syncthreads();
    for (int st = 128; st > 0; st >>= 1) {
        if (tid < st) red[tid] += red[tid + st];
        __syncthreads();
    }
    const float rstd = rsqrtf(red[0] * (1.0f / DIM_) + 1e-5f);

    #pragma unroll
    for (int i = 0; i < 4; ++i) {
        const int c = tid + i * 256;
        out[row * DIM_ + c] = (bf16_t)((v[i] - mean) * rstd * g[c] + b[c]);
    }
}

// ---------------------------------------------------------------------------
// Elementwise helpers
// ---------------------------------------------------------------------------
__global__ void __launch_bounds__(256)
cast_f32_to_bf16(const float* __restrict__ in, bf16_t* __restrict__ out, size_t n)
{
    const size_t i = (size_t)blockIdx.x * 256 + threadIdx.x;
    if (i < n) out[i] = (bf16_t)in[i];
}

__global__ void __launch_bounds__(256)
add_inplace(float* __restrict__ x, const float* __restrict__ t, size_t n)
{
    const size_t i = (size_t)blockIdx.x * 256 + threadIdx.x;
    if (i < n) x[i] += t[i];
}

__global__ void __launch_bounds__(256)
copy_f32(const float* __restrict__ in, float* __restrict__ out, size_t n)
{
    const size_t i = (size_t)blockIdx.x * 256 + threadIdx.x;
    if (i < n) out[i] = in[i];
}

// ---------------------------------------------------------------------------
// Feature maps on the 4 global heads (cols 512..1023):
//   q' = softmax(q, dim=e) ; k' = elu(k)+1. One wave (32 lanes) per row of 128.
// ---------------------------------------------------------------------------
__global__ void __launch_bounds__(256)
qk_featuremap(float* __restrict__ Q, float* __restrict__ K)
{
    const int wave = threadIdx.x >> 5, lane = threadIdx.x & 31;
    const int rr = blockIdx.x * 8 + wave;        // 0..32767
    const int bt = rr >> 2, hh = rr & 3;
    const size_t base = (size_t)bt * DIM_ + (size_t)(4 + hh) * DH_;

    float q[4];
    float mx = -1e30f;
    #pragma unroll
    for (int j = 0; j < 4; ++j) { q[j] = Q[base + lane + j * 32]; mx = fmaxf(mx, q[j]); }
    #pragma unroll
    for (int o = 16; o > 0; o >>= 1) mx = fmaxf(mx, __shfl_xor(mx, o, 32));
    float s = 0.0f;
    #pragma unroll
    for (int j = 0; j < 4; ++j) { q[j] = __expf(q[j] - mx); s += q[j]; }
    #pragma unroll
    for (int o = 16; o > 0; o >>= 1) s += __shfl_xor(s, o, 32);
    const float inv = 1.0f / s;
    #pragma unroll
    for (int j = 0; j < 4; ++j) Q[base + lane + j * 32] = q[j] * inv;

    #pragma unroll
    for (int j = 0; j < 4; ++j) {
        const float k = K[base + lane + j * 32];
        K[base + lane + j * 32] = (k > 0.0f) ? (k + 1.0f) : __expf(k);
    }
}

// ---------------------------------------------------------------------------
// Local windowed causal attention (heads 0..3), WINDOW=128, look_backward=1.
// grid (32 windows, 8 = b*4 heads), block 128 threads, thread = one query row.
// Flash-style online softmax; output accumulated in LDS (96KB total).
// ---------------------------------------------------------------------------
__global__ void __launch_bounds__(128)
local_attn_kernel(const float* __restrict__ Q, const float* __restrict__ K,
                  const float* __restrict__ V, float* __restrict__ O)
{
    __shared__ float os[128 * 128];  // 64KB output accumulator
    __shared__ float ks[32 * 128];   // 16KB key tile
    __shared__ float vs[32 * 128];   // 16KB value tile
    __shared__ int   tkv[32];

    const int i  = threadIdx.x;       // query row in window
    const int w  = blockIdx.x;        // window 0..31
    const int bh = blockIdx.y;        // 0..7
    const int bidx = bh >> 2, h = bh & 3;
    const int tq = w * 128 + i;
    const size_t rowQ = ((size_t)bidx * T_ + tq) * DIM_ + (size_t)h * DH_;

    for (int e = 0; e < 128; ++e) os[i * 128 + e] = 0.0f;
    float m = -1e30f, l = 0.0f;
    const float scale = 0.08838834764831845f;   // 1/sqrt(128)

    for (int jt = 0; jt < 8; ++jt) {
        __syncthreads();
        // cooperative tile load (32 keys x 128 dims)
        for (int c = 0; c < 32; ++c) {
            const int f = i * 32 + c;
            const int jj = f >> 7, e = f & 127;
            const int jg = jt * 32 + jj;       // 0..255 within look-around
            int tk; bool valid;
            if (jg < 128) { tk = (w - 1) * 128 + jg; valid = (w > 0); }
            else          { tk = w * 128 + (jg - 128); valid = true;  }
            float kv = 0.0f, vv = 0.0f;
            if (valid) {
                const size_t ro = ((size_t)bidx * T_ + tk) * DIM_ + (size_t)h * DH_ + e;
                kv = K[ro]; vv = V[ro];
            }
            ks[jj * 128 + e] = kv; vs[jj * 128 + e] = vv;
            if (e == 0) tkv[jj] = valid ? tk : -1;
        }
        __syncthreads();

        // dots for this tile (q row streamed from global in 32-wide chunks)
        float s[32];
        #pragma unroll
        for (int c = 0; c < 32; ++c) s[c] = 0.0f;
        for (int ec = 0; ec < 4; ++ec) {
            float qv[32];
            #pragma unroll
            for (int e2 = 0; e2 < 32; ++e2) qv[e2] = Q[rowQ + ec * 32 + e2];
            for (int c = 0; c < 32; ++c) {
                float d = 0.0f;
                #pragma unroll
                for (int e2 = 0; e2 < 32; ++e2) d += qv[e2] * ks[c * 128 + ec * 32 + e2];
                s[c] += d;
            }
        }
        float tmax = -1e30f;
        #pragma unroll
        for (int c = 0; c < 32; ++c) {
            const int tk = tkv[c];
            float d = s[c] * scale;
            if (tk < 0 || tk > tq) d = -1e30f;
            s[c] = d;
            tmax = fmaxf(tmax, d);
        }
        const float nm = fmaxf(m, tmax);
        const float alpha = __expf(m - nm);
        float ps = 0.0f;
        #pragma unroll
        for (int c = 0; c < 32; ++c) {
            s[c] = (s[c] <= -1e29f) ? 0.0f : __expf(s[c] - nm);
            ps += s[c];
        }
        l = l * alpha + ps;
        m = nm;
        for (int e = 0; e < 128; ++e) {
            float a = os[i * 128 + e] * alpha;
            #pragma unroll
            for (int c = 0; c < 32; ++c) a += s[c] * vs[c * 128 + e];
            os[i * 128 + e] = a;
        }
    }
    const float inv = 1.0f / l;
    for (int e = 0; e < 128; ++e) O[rowQ + e] = os[i * 128 + e] * inv;
}

// ---------------------------------------------------------------------------
// Linear attention (heads 4..7), BUCKET=64, u=64 buckets.
// G2: per-bucket ctx[d,e] = sum_n k'[n,d] v[n,e]  and ksum[d].
// ---------------------------------------------------------------------------
__global__ void __launch_bounds__(256)
linattn_bucket_ctx(const float* __restrict__ K, const float* __restrict__ V,
                   float* __restrict__ CTX, float* __restrict__ KS)
{
    __shared__ float ks[64 * 128];  // 32KB
    __shared__ float vs[64 * 128];  // 32KB
    const int u = blockIdx.x, bh = blockIdx.y;
    const int bidx = bh >> 2, hh = bh & 3;
    const int t0 = u * 64;

    for (int c = 0; c < 32; ++c) {
        const int f = threadIdx.x * 32 + c;   // 0..8191
        const int n = f >> 7, e = f & 127;
        const size_t ro = ((size_t)bidx * T_ + t0 + n) * DIM_ + (size_t)(4 + hh) * DH_ + e;
        ks[n * 128 + e] = K[ro];
        vs[n * 128 + e] = V[ro];
    }
    __syncthreads();

    if (threadIdx.x < 128) {
        const int d = threadIdx.x;
        float s = 0.0f;
        for (int n = 0; n < 64; ++n) s += ks[n * 128 + d];
        KS[((size_t)bh * 64 + u) * 128 + d] = s;
    }

    const int d  = threadIdx.x >> 1;
    const int e0 = (threadIdx.x & 1) * 64;
    float acc[64];
    #pragma unroll
    for (int e = 0; e < 64; ++e) acc[e] = 0.0f;
    for (int n = 0; n < 64; ++n) {
        const float kd = ks[n * 128 + d];
        #pragma unroll
        for (int e = 0; e < 64; ++e) acc[e] += kd * vs[n * 128 + e0 + e];
    }
    const size_t base = ((size_t)bh * 64 + u) * 16384 + (size_t)d * 128 + e0;
    for (int e = 0; e < 64; ++e) CTX[base + e] = acc[e];
}

// G3: cumsum over buckets + normalize by k_cum[d] + shift one bucket (blindspot).
__global__ void __launch_bounds__(256)
linattn_cumshift(float* __restrict__ CTX, const float* __restrict__ KS)
{
    const int bh = blockIdx.y;
    const int f = blockIdx.x * 256 + threadIdx.x;  // 0..16383 -> (d,e)
    const int d = f >> 7;
    float kacc = 0.0f, cacc = 0.0f;
    for (int u = 0; u < 64; ++u) {
        const size_t cb = ((size_t)bh * 64 + u) * 16384 + f;
        const float raw  = CTX[cb];
        const float kraw = KS[((size_t)bh * 64 + u) * 128 + d];
        CTX[cb] = (u == 0) ? 0.0f : cacc / (kacc + 1e-6f);
        cacc += raw;
        kacc += kraw;
    }
}

// G4: out[n,e] = sum_d q'[n,d] * ctx_shifted[d,e] per bucket.
__global__ void __launch_bounds__(256)
linattn_out(const float* __restrict__ Q, const float* __restrict__ CTX,
            float* __restrict__ O)
{
    __shared__ float cs[16384];     // 64KB ctx bucket
    __shared__ float qs[64 * 128];  // 32KB q' bucket
    const int u = blockIdx.x, bh = blockIdx.y;
    const int bidx = bh >> 2, hh = bh & 3;
    const size_t cb = ((size_t)bh * 64 + u) * 16384;

    for (int c = 0; c < 64; ++c)
        cs[threadIdx.x * 64 + c] = CTX[cb + threadIdx.x * 64 + c];
    for (int c = 0; c < 32; ++c) {
        const int f = threadIdx.x * 32 + c;
        const int n = f >> 7, e = f & 127;
        qs[n * 128 + e] = Q[((size_t)bidx * T_ + u * 64 + n) * DIM_ + (size_t)(4 + hh) * DH_ + e];
    }
    __syncthreads();

    const int n  = threadIdx.x >> 2;
    const int e0 = (threadIdx.x & 3) * 32;
    float acc[32];
    #pragma unroll
    for (int e = 0; e < 32; ++e) acc[e] = 0.0f;
    for (int d = 0; d < 128; ++d) {
        const float qv = qs[n * 128 + d];
        #pragma unroll
        for (int e = 0; e < 32; ++e) acc[e] += qv * cs[d * 128 + e0 + e];
    }
    const size_t ro = ((size_t)bidx * T_ + u * 64 + n) * DIM_ + (size_t)(4 + hh) * DH_ + e0;
    for (int e = 0; e < 32; ++e) O[ro + e] = acc[e];
}

// ---------------------------------------------------------------------------
// Host-side orchestration
// ---------------------------------------------------------------------------
static inline size_t alignup(size_t x) { return (x + 255) & ~(size_t)255; }

extern "C" void kernel_launch(void* const* d_in, const int* in_sizes, int n_in,
                              void* d_out, int out_size, void* d_ws, size_t ws_size,
                              hipStream_t stream)
{
    (void)in_sizes; (void)n_in; (void)out_size; (void)ws_size;

    const float* x    = (const float*)d_in[0];
    const float* Wq   = (const float*)d_in[1];
    const float* Wk   = (const float*)d_in[2];
    const float* Wv   = (const float*)d_in[3];
    const float* Wo   = (const float*)d_in[4];
    const float* bo   = (const float*)d_in[5];
    const float* ln1g = (const float*)d_in[6];
    const float* ln1b = (const float*)d_in[7];
    const float* W1   = (const float*)d_in[8];
    const float* b1   = (const float*)d_in[9];
    const float* W2   = (const float*)d_in[10];
    const float* b2   = (const float*)d_in[11];
    const float* ln2g = (const float*)d_in[12];
    const float* ln2b = (const float*)d_in[13];

    // ---- workspace layout (aliased; ~216MB total) ----
    char* ws = (char*)d_ws;
    size_t off = 0;
    auto take = [&](size_t bytes) { char* p = ws + off; off += alignup(bytes); return p; };

    float*  X   = (float*)take(NELEM * 4);                    // 32MB residual stream
    bf16_t* Hs  = (bf16_t*)take(NELEM * 2);                   // 16MB bf16 GEMM input (1024-wide)
    bf16_t* Hb  = (bf16_t*)take((size_t)MTOT * 4096 * 2);     // 64MB FFN hidden (bf16)
    float*  Q   = (float*)Hb;                                  // alias (dead before FFN1)
    float*  Kp  = (float*)((char*)Hb + NELEM * 4);             // alias, second half
    float*  V   = (float*)take(NELEM * 4);                    // 32MB
    float*  Aat = (float*)take(NELEM * 4);                    // 32MB attention output
    bf16_t* Wbf = (bf16_t*)take((size_t)4096 * 1024 * 2);     // 8MB weight cast scratch
    float*  Tmp = (float*)take(NELEM * 4);                    // 32MB GEMM f32 out
    float*  CTX = Tmp;                                         // alias (dead before Wo proj)
    float*  KS  = (float*)take((size_t)8 * 64 * 128 * 4);     // 256KB

    const size_t nElem   = NELEM;
    const int    blkCopy = (int)((nElem + 255) / 256);
    const dim3   gemmGrid1024(1024 / GTN, MTOT / GTM);   // N=1024
    const dim3   gemmGrid4096(4096 / GTN, MTOT / GTM);   // N=4096

    copy_f32<<<blkCopy, 256, 0, stream>>>(x, X, nElem);

    for (int l = 0; l < 4; ++l) {
        const size_t wOff  = (size_t)l * 1024 * 1024;
        const size_t w1Off = (size_t)l * 1024 * 4096;
        const size_t vOff  = (size_t)l * 1024;
        const size_t v1Off = (size_t)l * 4096;

        // ---- LN1 -> bf16 ----
        layernorm_to_bf16<<<MTOT, 256, 0, stream>>>(X, ln1g + vOff, ln1b + vOff, Hs);

        // ---- Q, K, V projections (WMMA) ----
        cast_f32_to_bf16<<<(1024 * 1024) / 256, 256, 0, stream>>>(Wq + wOff, Wbf, (size_t)1024 * 1024);
        gemm_bf16_wmma<0, 0><<<gemmGrid1024, 256, 0, stream>>>(Hs, Wbf, nullptr, Q, (bf16_t*)nullptr, MTOT, 1024, 1024);
        cast_f32_to_bf16<<<(1024 * 1024) / 256, 256, 0, stream>>>(Wk + wOff, Wbf, (size_t)1024 * 1024);
        gemm_bf16_wmma<0, 0><<<gemmGrid1024, 256, 0, stream>>>(Hs, Wbf, nullptr, Kp, (bf16_t*)nullptr, MTOT, 1024, 1024);
        cast_f32_to_bf16<<<(1024 * 1024) / 256, 256, 0, stream>>>(Wv + wOff, Wbf, (size_t)1024 * 1024);
        gemm_bf16_wmma<0, 0><<<gemmGrid1024, 256, 0, stream>>>(Hs, Wbf, nullptr, V, (bf16_t*)nullptr, MTOT, 1024, 1024);

        // ---- attention ----
        qk_featuremap<<<4096, 256, 0, stream>>>(Q, Kp);                      // global heads only
        local_attn_kernel<<<dim3(32, 8), 128, 0, stream>>>(Q, Kp, V, Aat);   // heads 0..3
        linattn_bucket_ctx<<<dim3(64, 8), 256, 0, stream>>>(Kp, V, CTX, KS); // heads 4..7
        linattn_cumshift<<<dim3(64, 8), 256, 0, stream>>>(CTX, KS);
        linattn_out<<<dim3(64, 8), 256, 0, stream>>>(Q, CTX, Aat);

        // ---- output projection + residual ----
        cast_f32_to_bf16<<<blkCopy, 256, 0, stream>>>(Aat, Hs, nElem);
        cast_f32_to_bf16<<<(1024 * 1024) / 256, 256, 0, stream>>>(Wo + wOff, Wbf, (size_t)1024 * 1024);
        gemm_bf16_wmma<0, 0><<<gemmGrid1024, 256, 0, stream>>>(Hs, Wbf, bo + vOff, Tmp, (bf16_t*)nullptr, MTOT, 1024, 1024);
        add_inplace<<<blkCopy, 256, 0, stream>>>(X, Tmp, nElem);

        // ---- FFN ----
        layernorm_to_bf16<<<MTOT, 256, 0, stream>>>(X, ln2g + vOff, ln2b + vOff, Hs);
        cast_f32_to_bf16<<<(1024 * 4096) / 256, 256, 0, stream>>>(W1 + w1Off, Wbf, (size_t)1024 * 4096);
        gemm_bf16_wmma<1, 1><<<gemmGrid4096, 256, 0, stream>>>(Hs, Wbf, b1 + v1Off, (float*)nullptr, Hb, MTOT, 4096, 1024);
        cast_f32_to_bf16<<<(1024 * 4096) / 256, 256, 0, stream>>>(W2 + w1Off, Wbf, (size_t)4096 * 1024);
        gemm_bf16_wmma<0, 0><<<gemmGrid1024, 256, 0, stream>>>(Hb, Wbf, b2 + vOff, Tmp, (bf16_t*)nullptr, MTOT, 1024, 4096);
        add_inplace<<<blkCopy, 256, 0, stream>>>(X, Tmp, nElem);
    }

    copy_f32<<<blkCopy, 256, 0, stream>>>(X, (float*)d_out, nElem);
}
